// MatrixModel_4226247819521
// MI455X (gfx1250) — matrix operand order repaired and checked by reading, not compiled, not run on hardware
//
#include <hip/hip_runtime.h>
#include <hip/hip_bf16.h>
#include <stdint.h>

// ---------------------------------------------------------------------------
// Model constants (reference: B=2,S=1024,V=16384,H=1024,L=2,NH=16,KV=4,KD=256,
// I=4096, head_dim=64, COARSE_ROWS=8192, top_k=64)
// ---------------------------------------------------------------------------
#define B_      2
#define S_      1024
#define H_      1024
#define KD_     256
#define I_      4096
#define V_      16384
#define L_      2
#define NROWS   (B_ * S_)        // 2048 token rows
#define COARSE_ 8192
#define TOPK_   64

typedef __bf16 bf16_t;
typedef __bf16 v4bf  __attribute__((ext_vector_type(4)));
typedef __bf16 v8bf  __attribute__((ext_vector_type(8)));
typedef __bf16 v16bf __attribute__((ext_vector_type(16)));
typedef float  v8f   __attribute__((ext_vector_type(8)));
typedef float  f32x4 __attribute__((ext_vector_type(4)));

union FragBF { v16bf v; v8bf h[2]; };

// ---------------------------------------------------------------------------
// CDNA5 async global->LDS copy helpers (ASYNCcnt-tracked, LDS written by the
// memory system, no VGPR round trip). LDS flat addresses carry the LDS byte
// offset in addr[31:0], so the low dword of the generic pointer is the vdst.
// The 24-bit instruction offset is added to BOTH the LDS and memory address.
// ---------------------------------------------------------------------------
__device__ __forceinline__ void async_copy16(const bf16_t* gsrc, bf16_t* ldst)
{
    unsigned lo = (unsigned)(uintptr_t)ldst;
    unsigned long long ga = (unsigned long long)(uintptr_t)gsrc;
    asm volatile("global_load_async_to_lds_b128 %0, %1, off"
                 :: "v"(lo), "v"(ga) : "memory");
}
__device__ __forceinline__ void async_copy32(const bf16_t* gsrc, bf16_t* ldst)
{
    unsigned lo = (unsigned)(uintptr_t)ldst;
    unsigned long long ga = (unsigned long long)(uintptr_t)gsrc;
    asm volatile("global_load_async_to_lds_b128 %0, %1, off"
                 :: "v"(lo), "v"(ga) : "memory");
    asm volatile("global_load_async_to_lds_b128 %0, %1, off offset:16"
                 :: "v"(lo), "v"(ga) : "memory");
}
__device__ __forceinline__ void wait_async0()
{
    asm volatile("s_wait_asynccnt 0x0" ::: "memory");
}

// ---------------------------------------------------------------------------
// GEMM: C[M,N] = A[M,K] * B[N,K]^T, A/B bf16 in global memory, C = f32 or bf16.
// M%128==0, N%128==0, K%32==0 at every call site. 128x128 macro tile,
// 256 threads = 8 waves (4M x 2N wave grid), 2x4 WMMA accumulators per wave.
// Both tiles staged with global_load_async_to_lds_b128 (32B per thread).
// LDS rows padded to 40 bf16 so all 16B fragment chunks stay 16B-aligned.
// ---------------------------------------------------------------------------
#define LDSP 40

template <typename TC>
__global__ __launch_bounds__(256) void gemm_bf16_async(
    const bf16_t* __restrict__ A, const bf16_t* __restrict__ Bm,
    TC* __restrict__ C, int K, int lda, int ldb, int ldc)
{
    __shared__ bf16_t Asub[128 * LDSP];
    __shared__ bf16_t Bsub[128 * LDSP];

    const int tid   = threadIdx.x;
    const int lane  = tid & 31;
    const int wid   = tid >> 5;
    const int waveM = wid & 3;
    const int waveN = wid >> 2;
    const int bm    = blockIdx.y * 128;
    const int bn    = blockIdx.x * 128;

    v8f acc[2][4] = {};

    const int ldRow = tid >> 1;        // 0..127
    const int ldCol = (tid & 1) * 16;  // 0 / 16 (elements)

    const bf16_t* aSrc = A  + (size_t)(bm + ldRow) * lda + ldCol;
    const bf16_t* bSrc = Bm + (size_t)(bn + ldRow) * ldb + ldCol;
    bf16_t* aDst = Asub + ldRow * LDSP + ldCol;
    bf16_t* bDst = Bsub + ldRow * LDSP + ldCol;

    for (int k0 = 0; k0 < K; k0 += 32) {
        // async-stage both tiles (no VALU, tracked by ASYNCcnt)
        async_copy32(aSrc + k0, aDst);
        async_copy32(bSrc + k0, bDst);
        wait_async0();
        __syncthreads();

        // fragment loads per ISA 16-bit WMMA VGPR layouts
        const int rsel = lane & 15;
        const int hsel = lane >> 4;
        FragBF afr[2], bfr[4];
        #pragma unroll
        for (int i = 0; i < 2; ++i) {
            const bf16_t* p = Asub + (waveM * 32 + i * 16 + rsel) * LDSP + hsel * 8;
            afr[i].h[0] = *(const v8bf*)(p);
            afr[i].h[1] = *(const v8bf*)(p + 16);
        }
        #pragma unroll
        for (int j = 0; j < 4; ++j) {
            const bf16_t* p = Bsub + (waveN * 64 + j * 16 + rsel) * LDSP + hsel * 8;
            bfr[j].h[0] = *(const v8bf*)(p);
            bfr[j].h[1] = *(const v8bf*)(p + 16);
        }
        #pragma unroll
        for (int i = 0; i < 2; ++i)
            #pragma unroll
            for (int j = 0; j < 4; ++j)
                acc[i][j] = __builtin_amdgcn_wmma_f32_16x16x32_bf16(
                    false, afr[i].v, false, bfr[j].v, (short)0, acc[i][j], false, false);
        __syncthreads();   // protect LDS reuse before next async issue
    }

    #pragma unroll
    for (int i = 0; i < 2; ++i) {
        const int rbase = bm + waveM * 32 + i * 16 + (lane >> 4) * 8;
        #pragma unroll
        for (int j = 0; j < 4; ++j) {
            const int c = bn + waveN * 64 + j * 16 + (lane & 15);
            #pragma unroll
            for (int e = 0; e < 8; ++e)
                C[(size_t)(rbase + e) * ldc + c] = (TC)acc[i][j][e];
        }
    }
}

template __global__ void gemm_bf16_async<bf16_t>(const bf16_t*, const bf16_t*, bf16_t*, int, int, int, int);
template __global__ void gemm_bf16_async<float >(const bf16_t*, const bf16_t*, float*,  int, int, int, int);

// ---------------------------------------------------------------------------
// Fused attention (faithful): scores = QK^T/8 via WMMA, softmax over 1024
// keys, factor = sum_q softmax, out = factor * v. q/k/v/out are bf16.
// Keys async-staged in 256-row chunks. Dynamic LDS ~= 102.3 KB.
// ---------------------------------------------------------------------------
#define ATT_SMEM_KT   (16 * 1024 * 4)
#define ATT_SMEM_QT   (ATT_SMEM_KT + 256 * 72 * 2)
#define ATT_SMEM_FAC  (ATT_SMEM_QT + 16 * 72 * 2)
#define ATT_SMEM_SIZE (ATT_SMEM_FAC + 64)

__global__ __launch_bounds__(256) void attn_kernel(
    const bf16_t* __restrict__ qb, const bf16_t* __restrict__ kb,
    const bf16_t* __restrict__ vb, bf16_t* __restrict__ outb)
{
    extern __shared__ char smem[];
    float*  scores = (float*)(smem);
    bf16_t* Kt     = (bf16_t*)(smem + ATT_SMEM_KT);
    bf16_t* Qt     = (bf16_t*)(smem + ATT_SMEM_QT);
    float*  factor = (float*)(smem + ATT_SMEM_FAC);

    const int t    = threadIdx.x;
    const int lane = t & 31;
    const int wid  = t >> 5;
    const int hidx = blockIdx.y;         // 0..31
    const int b    = hidx >> 4;
    const int kv   = (hidx >> 2) & 3;
    const int gi   = hidx & 3;
    const int s0   = blockIdx.x * 16;
    const size_t rowBase = (size_t)(b * S_ + s0);
    const int qoff = kv * 256 + gi * 64;
    const int koff = kv * 64;

    // async-stage Q tile 16x64 bf16 (threads 0..127, 16B each)
    if (t < 128) {
        const int r = t >> 3, c8 = (t & 7) * 8;
        async_copy16(qb + (rowBase + r) * H_ + qoff + c8, Qt + r * 72 + c8);
    }

    for (int ch = 0; ch < 4; ++ch) {
        const int n0 = ch * 256;
        // async-stage 256 key rows x 64 bf16 (one row = 128B per thread)
        {
            const bf16_t* src = kb + (size_t)(b * S_ + n0 + t) * KD_ + koff;
            bf16_t* dst = Kt + t * 72;
            #pragma unroll
            for (int i = 0; i < 64; i += 8) async_copy16(src + i, dst + i);
        }
        wait_async0();
        __syncthreads();

        #pragma unroll
        for (int jt = 0; jt < 2; ++jt) {
            const int ntile = wid * 2 + jt;
            v8f acc = {};
            #pragma unroll
            for (int kt = 0; kt < 2; ++kt) {
                FragBF a, bf;
                const bf16_t* pa = Qt + (lane & 15) * 72 + kt * 32 + (lane >> 4) * 8;
                a.h[0] = *(const v8bf*)(pa);
                a.h[1] = *(const v8bf*)(pa + 16);
                const bf16_t* pb = Kt + (ntile * 16 + (lane & 15)) * 72 + kt * 32 + (lane >> 4) * 8;
                bf.h[0] = *(const v8bf*)(pb);
                bf.h[1] = *(const v8bf*)(pb + 16);
                acc = __builtin_amdgcn_wmma_f32_16x16x32_bf16(
                    false, a.v, false, bf.v, (short)0, acc, false, false);
            }
            const int col = n0 + ntile * 16 + (lane & 15);
            #pragma unroll
            for (int e = 0; e < 8; ++e)
                scores[(e + (lane >> 4) * 8) * 1024 + col] = acc[e] * 0.125f;
        }
        __syncthreads();
    }

    // per-row softmax; factor = sum of probabilities (reference contracts only q)
    #pragma unroll
    for (int rr = 0; rr < 2; ++rr) {
        const int r = wid * 2 + rr;
        const float* srow = scores + r * 1024;
        float m = -__builtin_inff();
        for (int i = lane; i < 1024; i += 32) m = fmaxf(m, srow[i]);
        for (int off = 16; off > 0; off >>= 1) m = fmaxf(m, __shfl_xor(m, off, 32));
        float den = 0.f;
        for (int i = lane; i < 1024; i += 32) den += __expf(srow[i] - m);
        for (int off = 16; off > 0; off >>= 1) den += __shfl_xor(den, off, 32);
        float ps = 0.f;
        for (int i = lane; i < 1024; i += 32) ps += __expf(srow[i] - m) / den;
        for (int off = 16; off > 0; off >>= 1) ps += __shfl_xor(ps, off, 32);
        if (lane == 0) factor[r] = ps;
    }
    __syncthreads();

    // out[b,s, kv*256+gi*64+d] = factor[s] * v[b,s, kv*64+d]
    {
        const int r = t >> 4, c = (t & 15) * 4;
        v4bf vv = *(const v4bf*)(vb + (rowBase + r) * KD_ + koff + c);
        const float f = factor[r];
        v4bf o;
        #pragma unroll
        for (int j = 0; j < 4; ++j) o[j] = (bf16_t)((float)vv[j] * f);
        *(v4bf*)(outb + (rowBase + r) * H_ + qoff + c) = o;
    }
}

// ---------------------------------------------------------------------------
// Small kernels
// ---------------------------------------------------------------------------
__global__ __launch_bounds__(256) void f32_to_bf16_kernel(
    const float* __restrict__ src, bf16_t* __restrict__ dst)
{
    const size_t i = ((size_t)blockIdx.x * 256 + threadIdx.x) * 4;
    f32x4 f = *(const f32x4*)(src + i);
    v4bf o;
    o[0] = (bf16_t)f.x; o[1] = (bf16_t)f.y; o[2] = (bf16_t)f.z; o[3] = (bf16_t)f.w;
    *(v4bf*)(dst + i) = o;
}

__global__ __launch_bounds__(256) void gather_kernel(
    const int* __restrict__ ids, const float* __restrict__ embed, bf16_t* __restrict__ h)
{
    const int row = blockIdx.x;
    const int t   = threadIdx.x;
    f32x4 f = *(const f32x4*)(embed + (size_t)ids[row] * H_ + t * 4);
    v4bf o;
    o[0] = (bf16_t)f.x; o[1] = (bf16_t)f.y; o[2] = (bf16_t)f.z; o[3] = (bf16_t)f.w;
    *(v4bf*)(h + (size_t)row * H_ + t * 4) = o;
}

__global__ __launch_bounds__(256) void mul_kernel(
    bf16_t* __restrict__ g, const bf16_t* __restrict__ u)
{
    const size_t i = ((size_t)blockIdx.x * 256 + threadIdx.x) * 4;
    v4bf a = *(const v4bf*)(g + i);
    v4bf b = *(const v4bf*)(u + i);
    v4bf o;
    #pragma unroll
    for (int j = 0; j < 4; ++j) o[j] = (bf16_t)((float)a[j] * (float)b[j]);
    *(v4bf*)(g + i) = o;
}

// iterative top-64 per row; mutates coarse (scratch, recomputed every call).
// tie rule: lowest index wins (matches jax.lax.top_k).
__global__ __launch_bounds__(256) void topk_kernel(
    float* __restrict__ coarse, int* __restrict__ outIdx)
{
    __shared__ float bv[256];
    __shared__ int   bi[256];
    const int row = blockIdx.x;
    const int t   = threadIdx.x;
    float* c = coarse + (size_t)row * COARSE_;
    for (int it = 0; it < TOPK_; ++it) {
        float mv = -__builtin_inff();
        int   mi = 0x7fffffff;
        for (int i = t; i < COARSE_; i += 256) {
            const float v = c[i];
            if (v > mv) { mv = v; mi = i; }
        }
        bv[t] = mv; bi[t] = mi;
        __syncthreads();
        for (int s = 128; s > 0; s >>= 1) {
            if (t < s) {
                if (bv[t+s] > bv[t] || (bv[t+s] == bv[t] && bi[t+s] < bi[t])) {
                    bv[t] = bv[t+s]; bi[t] = bi[t+s];
                }
            }
            __syncthreads();
        }
        if (t == 0) {
            outIdx[row * TOPK_ + it] = bi[0];
            c[bi[0]] = -__builtin_inff();
        }
        __syncthreads();
    }
}

__global__ __launch_bounds__(256) void fine_logits_kernel(
    const bf16_t* __restrict__ h, const bf16_t* __restrict__ Wout,
    const int* __restrict__ idx, float* __restrict__ logits)
{
    __shared__ float part[256];
    const int row = blockIdx.x;
    const int t   = threadIdx.x;
    const int kk  = t >> 2;
    const int sub = t & 3;
    const bf16_t* hv = h + (size_t)row * H_;
    const int id = idx[row * TOPK_ + kk];
    const bf16_t* w = Wout + (size_t)id * H_;
    float s = 0.f;
    const int base = sub * 256;
    for (int i = base; i < base + 256; ++i) s += (float)hv[i] * (float)w[i];
    part[t] = s;
    __syncthreads();
    if (sub == 0)
        logits[row * TOPK_ + kk] = part[t] + part[t+1] + part[t+2] + part[t+3];
}

// ---------------------------------------------------------------------------
// Host orchestration
// ---------------------------------------------------------------------------
extern "C" void kernel_launch(void* const* d_in, const int* in_sizes, int n_in,
                              void* d_out, int out_size, void* d_ws, size_t ws_size,
                              hipStream_t stream)
{
    (void)in_sizes; (void)n_in; (void)out_size; (void)ws_size;

    const int*   ids   = (const int*)d_in[0];
    const float* embed = (const float*)d_in[1];
    const float* Wq    = (const float*)d_in[2];
    const float* Wk    = (const float*)d_in[3];
    const float* Wv    = (const float*)d_in[4];
    const float* Wo    = (const float*)d_in[5];
    const float* Wg    = (const float*)d_in[6];
    const float* Wu    = (const float*)d_in[7];
    const float* Wd    = (const float*)d_in[8];
    const float* Wout  = (const float*)d_in[9];
    // d_in[10] = top_k scalar (== 64, hardcoded)

    float* logits = (float*)d_out;                                   // [2048,64] f32
    int*   outIdx = (int*)((float*)d_out + (size_t)NROWS * TOPK_);   // [2048,64] i32

    // workspace: bf16 weight copies | bf16 activations | big region (coarse f32)
    bf16_t* p = (bf16_t*)d_ws;
    bf16_t* wq_bf = p;   p += (size_t)L_ * H_ * H_;
    bf16_t* wk_bf = p;   p += (size_t)L_ * KD_ * H_;
    bf16_t* wv_bf = p;   p += (size_t)L_ * KD_ * H_;
    bf16_t* wo_bf = p;   p += (size_t)L_ * H_ * H_;
    bf16_t* wg_bf = p;   p += (size_t)L_ * I_ * H_;
    bf16_t* wu_bf = p;   p += (size_t)L_ * I_ * H_;
    bf16_t* wd_bf = p;   p += (size_t)L_ * I_ * H_;
    bf16_t* wout_bf = p; p += (size_t)V_ * H_;
    bf16_t* h_bf  = p;   p += (size_t)NROWS * H_;
    bf16_t* h2_bf = p;   p += (size_t)NROWS * H_;
    bf16_t* big   = p;   // time-shared: {q,k,v} -> {gate,up} -> coarse(f32)
    bf16_t* q_bf  = big;
    bf16_t* k_bf  = big + (size_t)NROWS * H_;
    bf16_t* v_bf  = k_bf + (size_t)NROWS * KD_;
    bf16_t* gate_bf = big;
    bf16_t* up_bf   = big + (size_t)NROWS * I_;
    float*  coarse  = (float*)big;

    hipFuncSetAttribute(reinterpret_cast<const void*>(attn_kernel),
                        hipFuncAttributeMaxDynamicSharedMemorySize, ATT_SMEM_SIZE);

    auto cvt = [&](const float* s, bf16_t* d, size_t n) {
        f32_to_bf16_kernel<<<(unsigned)(n / 1024), 256, 0, stream>>>(s, d);
    };
    auto gemm = [&](const bf16_t* A, const bf16_t* Bm, bf16_t* C,
                    int M, int N, int K, int lda, int ldb, int ldc) {
        gemm_bf16_async<bf16_t><<<dim3(N / 128, M / 128), 256, 0, stream>>>(A, Bm, C, K, lda, ldb, ldc);
    };

    // one-time (per call) bf16 weight conversion
    cvt(Wq,   wq_bf,   (size_t)L_ * H_ * H_);
    cvt(Wk,   wk_bf,   (size_t)L_ * KD_ * H_);
    cvt(Wv,   wv_bf,   (size_t)L_ * KD_ * H_);
    cvt(Wo,   wo_bf,   (size_t)L_ * H_ * H_);
    cvt(Wg,   wg_bf,   (size_t)L_ * I_ * H_);
    cvt(Wu,   wu_bf,   (size_t)L_ * I_ * H_);
    cvt(Wd,   wd_bf,   (size_t)L_ * I_ * H_);
    cvt(Wout, wout_bf, (size_t)V_ * H_);

    gather_kernel<<<NROWS, 256, 0, stream>>>(ids, embed, h_bf);

    for (int l = 0; l < L_; ++l) {
        const bf16_t* wq = wq_bf + (size_t)l * H_ * H_;
        const bf16_t* wk = wk_bf + (size_t)l * KD_ * H_;
        const bf16_t* wv = wv_bf + (size_t)l * KD_ * H_;
        const bf16_t* wo = wo_bf + (size_t)l * H_ * H_;
        const bf16_t* wg = wg_bf + (size_t)l * I_ * H_;
        const bf16_t* wu = wu_bf + (size_t)l * I_ * H_;
        const bf16_t* wd = wd_bf + (size_t)l * H_ * I_;

        gemm(h_bf, wq, q_bf, NROWS, H_,  H_, H_, H_, H_);
        gemm(h_bf, wk, k_bf, NROWS, KD_, H_, H_, H_, KD_);
        gemm(h_bf, wv, v_bf, NROWS, KD_, H_, H_, H_, KD_);

        attn_kernel<<<dim3(S_ / 16, 32), 256, ATT_SMEM_SIZE, stream>>>(q_bf, k_bf, v_bf, h2_bf);

        gemm(h2_bf, wo, h_bf,  NROWS, H_, H_, H_, H_, H_);
        gemm(h_bf,  wg, gate_bf, NROWS, I_, H_, H_, H_, I_);
        gemm(h_bf,  wu, up_bf,   NROWS, I_, H_, H_, H_, I_);
        mul_kernel<<<(unsigned)(((size_t)NROWS * I_) / 1024), 256, 0, stream>>>(gate_bf, up_bf);
        gemm(gate_bf, wd, h_bf, NROWS, H_, I_, I_, I_, H_);
    }

    // coarse logits (f32 out) -> top-64 -> fine per-index row dots
    gemm_bf16_async<float><<<dim3(COARSE_ / 128, NROWS / 128), 256, 0, stream>>>(
        h_bf, wout_bf, coarse, H_, H_, H_, COARSE_);
    topk_kernel<<<NROWS, 256, 0, stream>>>(coarse, outIdx);
    fine_logits_kernel<<<NROWS, 256, 0, stream>>>(h_bf, wout_bf, outIdx, logits);
}
